// SigmoidScaledDotProductAttention_13460427505966
// MI455X (gfx1250) — matrix operand order repaired and checked by reading, not compiled, not run on hardware
//
#include <hip/hip_runtime.h>
#include <hip/hip_bf16.h>

typedef __attribute__((ext_vector_type(16))) _Float16 v16h;
typedef __attribute__((ext_vector_type(8)))  float    v8f;
typedef __attribute__((ext_vector_type(4)))  float    v4f;

#define ATT_B 4
#define ATT_H 16
#define ATT_L 2048
#define ATT_D 64

constexpr int KC  = 32;   // attn columns processed per iteration
constexpr int SKH = 72;   // LDS row stride (halves) for f16 K tile  [KC][D]
constexpr int VTH = 40;   // LDS row stride (halves) for f16 V^T tile [D][KC]
constexpr int AW  = 40;   // LDS row stride (halves) for per-wave attn staging [16][KC]

// 16 consecutive f16 in LDS -> v16h (two ds_load_b128, no VALU)
__device__ __forceinline__ v16h lds_h16(const _Float16* p) {
  v16h r;
  ((v4f*)&r)[0] = *(const v4f*)(p);
  ((v4f*)&r)[1] = *(const v4f*)(p + 8);
  return r;
}

__global__ __launch_bounds__(256)
void sigmoid_attn_wmma_kernel(const float* __restrict__ Q,
                              const float* __restrict__ K,
                              const float* __restrict__ V,
                              const int*   __restrict__ M,
                              float* __restrict__ out,
                              float* __restrict__ attn) {
  __shared__ __align__(16) _Float16 sKh[KC * SKH];      // K chunk, f16 row-major
  __shared__ __align__(16) _Float16 sVth[ATT_D * VTH];  // V chunk, f16 transposed
  __shared__ __align__(16) _Float16 sA[8 * 16 * AW];    // per-wave attn transpose stage

  const int tid  = threadIdx.x;
  const int wave = tid >> 5;
  const int lane = tid & 31;
  const int half = lane >> 4;   // which 16-lane half of the wave
  const int l16  = lane & 15;

  const int bh = blockIdx.x >> 4;              // 16 blocks per (b,h) slice
  const int b  = bh >> 4;                      // H == 16
  const int q0 = ((blockIdx.x & 15) * 8 + wave) * 16;

  const float* Qs = Q + (size_t)bh * ATT_L * ATT_D;
  const float* Ks = K + (size_t)bh * ATT_L * ATT_D;
  const float* Vs = V + (size_t)bh * ATT_L * ATT_D;
  const int*   Ms = M + (size_t)b  * ATT_L * ATT_L;
  float* outS  = out  + (size_t)bh * ATT_L * ATT_D;
  float* attnS = attn + (size_t)bh * ATT_L * ATT_L;

  // ---- Q fragments (f16 A-matrix 16x32 layout), pre-scaled by 1/8 ----------
  // A layout: lanes0-15 row M=lane, halves 0..7 <-> K=0..7, halves 8..15 <-> K=16..23
  //           lanes16-31 row M=lane-16, K=8..15 and K=24..31
  v16h aQ[2];
  {
    const float* qr = Qs + (size_t)(q0 + l16) * ATT_D;
#pragma unroll
    for (int j = 0; j < 2; ++j) {
      const int d0 = 32 * j + half * 8;
#pragma unroll
      for (int gi = 0; gi < 2; ++gi) {
        const v4f* p4 = (const v4f*)(qr + d0 + 16 * gi);
#pragma unroll
        for (int g = 0; g < 2; ++g) {
          v4f t = p4[g];
#pragma unroll
          for (int i = 0; i < 4; ++i)
            aQ[j][8 * gi + 4 * g + i] = (_Float16)(t[i] * 0.125f);
        }
      }
    }
  }

  v8f acc[4];
#pragma unroll
  for (int dt = 0; dt < 4; ++dt)
#pragma unroll
    for (int i = 0; i < 8; ++i) acc[dt][i] = 0.0f;

  const int rowSt = tid >> 3;          // staging row within K/V tile (0..31)
  const int colSt = (tid & 7) * 8;     // staging first column (0..56)

  for (int k0 = 0; k0 < ATT_L; k0 += KC) {
    // ---- cooperative staging with f32->f16 conversion done ONCE per block ---
    {
      const float* kg = Ks + (size_t)(k0 + rowSt) * ATT_D + colSt;
      const float* vg = Vs + (size_t)(k0 + rowSt) * ATT_D + colSt;
      v4f k0v = ((const v4f*)kg)[0], k1v = ((const v4f*)kg)[1];
      v4f v0v = ((const v4f*)vg)[0], v1v = ((const v4f*)vg)[1];

      _Float16 kh[8];
#pragma unroll
      for (int i = 0; i < 4; ++i) {
        kh[i]     = (_Float16)k0v[i];
        kh[4 + i] = (_Float16)k1v[i];
      }
      *(v4f*)&sKh[rowSt * SKH + colSt] = *(const v4f*)kh;   // 8 halves = 16B

#pragma unroll
      for (int i = 0; i < 4; ++i) {                          // transposed f16 V
        sVth[(colSt + i)     * VTH + rowSt] = (_Float16)v0v[i];
        sVth[(colSt + i + 4) * VTH + rowSt] = (_Float16)v1v[i];
      }

      if (k0 + KC < ATT_L) {             // next chunk -> global_prefetch_b8
        __builtin_prefetch(kg + (size_t)KC * ATT_D, 0, 0);
        __builtin_prefetch(vg + (size_t)KC * ATT_D, 0, 0);
      }
    }
    __syncthreads();

    // ---- scores S(16x32) = (Q/8) . K^T : two 16x16 C tiles ------------------
#pragma unroll
    for (int h = 0; h < 2; ++h) {
      v8f c = {};
#pragma unroll
      for (int j = 0; j < 2; ++j) {
        // B(d,n): column n = l16 -> K-row (h*16+l16); contraction d = 32j + half*8 + {0..7, 16..23}
        v16h bK;
        {
          const _Float16* pk = &sKh[(h * 16 + l16) * SKH + 32 * j + half * 8];
          ((v4f*)&bK)[0] = *(const v4f*)(pk);
          ((v4f*)&bK)[1] = *(const v4f*)(pk + 16);
        }
        c = __builtin_amdgcn_wmma_f32_16x16x32_f16(false, aQ[j], false, bK,
                                                   (short)0, c, false, false);
      }
      // C layout: vgpr r, lanes0-15: M=r ; lanes16-31: M=r+8 ; N = l16
      _Float16* aRow = &sA[(wave * 16) * AW];
#pragma unroll
      for (int r = 0; r < 8; ++r) {
        const int qg = q0 + r + half * 8;
        const int kg = k0 + h * 16 + l16;
        const float s = c[r];
        const int   mv = Ms[(size_t)qg * ATT_L + kg];
        const float a  = mv ? __builtin_amdgcn_rcpf(1.0f + __expf(-s)) : 0.0f;
        __builtin_nontemporal_store(a, &attnS[(size_t)qg * ATT_L + kg]);
        aRow[(r + half * 8) * AW + h * 16 + l16] = (_Float16)a;  // transpose stage
      }
    }

    // ---- A fragment of attn tile (f16 16x32) from per-wave LDS --------------
    v16h aA;
    {
      const _Float16* pa = &sA[(wave * 16 + l16) * AW + half * 8];
      ((v4f*)&aA)[0] = *(const v4f*)(pa);        // K 0..7   (or 8..15)
      ((v4f*)&aA)[1] = *(const v4f*)(pa + 16);   // K 16..23 (or 24..31)
    }

    // ---- PV: out(16x64) += attn(16x32) x V(32x64) ---------------------------
#pragma unroll
    for (int dt = 0; dt < 4; ++dt) {
      // B(kk,n): column n = l16 -> output dim d = 16*dt + l16 ; kk = half*8 + {0..7, 16..23}
      v16h bV;
      {
        const _Float16* pv = &sVth[(16 * dt + l16) * VTH + half * 8];
        ((v4f*)&bV)[0] = *(const v4f*)(pv);
        ((v4f*)&bV)[1] = *(const v4f*)(pv + 16);
      }
      acc[dt] = __builtin_amdgcn_wmma_f32_16x16x32_f16(false, aA, false, bV,
                                                       (short)0, acc[dt], false, false);
    }

    __syncthreads();
  }

  // ---- write 16x64 output tile ---------------------------------------------
#pragma unroll
  for (int dt = 0; dt < 4; ++dt)
#pragma unroll
    for (int r = 0; r < 8; ++r) {
      const int qg = q0 + r + half * 8;
      __builtin_nontemporal_store(acc[dt][r],
                                  &outS[(size_t)qg * ATT_D + dt * 16 + l16]);
    }
}

extern "C" void kernel_launch(void* const* d_in, const int* in_sizes, int n_in,
                              void* d_out, int out_size, void* d_ws, size_t ws_size,
                              hipStream_t stream) {
  (void)in_sizes; (void)n_in; (void)out_size; (void)d_ws; (void)ws_size;
  const float* Q = (const float*)d_in[0];
  const float* K = (const float*)d_in[1];
  const float* V = (const float*)d_in[2];
  const int*   M = (const int*)d_in[3];
  float* out  = (float*)d_out;
  float* attn = out + (size_t)ATT_B * ATT_H * ATT_L * ATT_D;

  const int qtiles = ATT_B * ATT_H * (ATT_L / 16);  // 8192
  dim3 grid(qtiles / 8);                            // 8 waves (q-tiles) per block
  dim3 block(256);
  sigmoid_attn_wmma_kernel<<<grid, block, 0, stream>>>(Q, K, V, M, out, attn);
}
